// Seq2Seq_52226802319758
// MI455X (gfx1250) — matrix operand order repaired and checked by reading, not compiled, not run on hardware
//
#include <hip/hip_runtime.h>
#include <hip/hip_bf16.h>
#include <cstdint>

typedef __attribute__((ext_vector_type(16))) __bf16 v16bf;
typedef __attribute__((ext_vector_type(8)))  __bf16 v8bf;
typedef __attribute__((ext_vector_type(8)))  float  v8f;

#define NH   896          // hidden
#define NE   300          // embedding
#define NEP  320          // embedding padded to mult of 32 (K for WMMA)
#define NV   30000        // vocab
#define NB   64           // batch
#define TE   40
#define TDD  40
#define G3   (3*NH)       // 2688
#define HH2  (2*NH)       // 1792

#if __has_builtin(__builtin_amdgcn_global_load_async_to_lds_b128)
#  define HAVE_ASYNC_LDS 1
#endif

// b128 async copy operand types (signature leaked by round-2 diagnostic:
// param is 'int __attribute__((vector_size(16))) addrspace-qualified *')
typedef int vint4_t __attribute__((vector_size(16)));
typedef __attribute__((address_space(1))) vint4_t* gptr_b128;
typedef __attribute__((address_space(3))) vint4_t* lptr_b128;

// 16-byte global -> LDS copy: async DMA path on gfx1250, ds_store fallback
__device__ __forceinline__ void cp16_g2l(const void* g, void* l){
#if defined(HAVE_ASYNC_LDS)
  __builtin_amdgcn_global_load_async_to_lds_b128((gptr_b128)g, (lptr_b128)l, 0, 0);
#else
  *(uint4*)l = *(const uint4*)g;
#endif
}
__device__ __forceinline__ void async_wait0(){
#if __has_builtin(__builtin_amdgcn_s_wait_asynccnt)
  __builtin_amdgcn_s_wait_asynccnt(0);
#elif defined(HAVE_ASYNC_LDS)
  asm volatile("s_wait_asynccnt 0" ::: "memory");
#endif
}

__device__ __forceinline__ unsigned short f32_to_bf16(float f) {
  unsigned int u = __float_as_uint(f);
  u += 0x7FFFu + ((u >> 16) & 1u);     // round to nearest even
  return (unsigned short)(u >> 16);
}
__device__ __forceinline__ float sigmoidf_(float x){ return 1.0f/(1.0f+__expf(-x)); }

// ---------------------------------------------------------------- utilities
__global__ void k_zero_f32(float* p, int n){
  int i = blockIdx.x*blockDim.x + threadIdx.x;
  if (i < n) p[i] = 0.f;
}

// f32 [Nrows,Ksrc] -> bf16 [Nrows,Kdst] (zero-padded K)
__global__ void k_conv_bf16(const float* __restrict__ src, unsigned short* __restrict__ dst,
                            int Nrows, int Ksrc, int Kdst){
  long long i = (long long)blockIdx.x*blockDim.x + threadIdx.x;
  long long tot = (long long)Nrows*Kdst;
  if (i >= tot) return;
  int n = (int)(i / Kdst), k = (int)(i % Kdst);
  float v = (k < Ksrc) ? src[(long long)n*Ksrc + k] : 0.f;
  dst[i] = f32_to_bf16(v);
}

// xs[t,b,:NEP] = bf16(emb[x[b,t],:NE]) zero padded
__global__ void k_gather_emb(const float* __restrict__ emb, const int* __restrict__ xidx,
                             unsigned short* __restrict__ dst, int T){
  long long i = (long long)blockIdx.x*blockDim.x + threadIdx.x;
  long long tot = (long long)T*NB*NEP;
  if (i >= tot) return;
  int t   = (int)(i / (NB*NEP));
  int rem = (int)(i % (NB*NEP));
  int b = rem / NEP, k = rem % NEP;
  int tok = xidx[b*T + t];
  float v = (k < NE) ? emb[(long long)tok*NE + k] : 0.f;
  dst[i] = f32_to_bf16(v);
}

// ------------------------------------------------------------ WMMA fragments
// A (16x32 bf16, MxK): lane<16 -> row=lane, K {0..15}; lane>=16 -> K {16..31}
__device__ __forceinline__ v16bf load_a16x32(const __bf16* __restrict__ A, int K,
                                             int row, int k, int koffA){
  v8bf lo = *(const v8bf*)(A + (size_t)row*K + k + koffA);
  v8bf hi = *(const v8bf*)(A + (size_t)row*K + k + 8 + koffA);
  v16bf a;
#pragma unroll
  for (int i=0;i<8;i++){ a[i]=lo[i]; a[8+i]=hi[i]; }
  return a;
}

// C = A[M,K]bf16 @ W[N,K]bf16^T + bias   (one wave per 16x16 tile)
__global__ __launch_bounds__(256) void k_wmma_gemm_tile(
    const unsigned short* __restrict__ A_, const unsigned short* __restrict__ W_,
    const float* __restrict__ bias, float* __restrict__ C, int M, int N, int K)
{
  const __bf16* A = (const __bf16*)A_;
  const __bf16* W = (const __bf16*)W_;
  int wave = (int)((blockIdx.x * 256 + threadIdx.x) >> 5);
  int lane = threadIdx.x & 31;
  int tn_cnt = N >> 4;
  int total  = (M >> 4) * tn_cnt;
  if (wave >= total) return;
  int tm = wave / tn_cnt, tn = wave - tm*tn_cnt;
  int l15   = lane & 15;
  int koffA = (lane >= 16) ? 16 : 0;
  int koffB = (lane >= 16) ? 16 : 0;
  int arow  = tm*16 + l15;
  const __bf16* wrow = W + (size_t)(tn*16 + l15)*K;   // B col = W row
  v8f acc = {};
  for (int k = 0; k < K; k += 32){
    v16bf b = *(const v16bf*)(wrow + k + koffB);
    v16bf a = load_a16x32(A, K, arow, k, koffA);
    acc = __builtin_amdgcn_wmma_f32_16x16x32_bf16(false, a, false, b,
                                                  (short)0, acc, false, false);
  }
  int col = tn*16 + l15;
  float bv = bias ? bias[col] : 0.f;
  int r0 = tm*16 + ((lane>=16)?8:0);
#pragma unroll
  for (int j=0;j<8;j++)
    C[(size_t)(r0+j)*N + col] = acc[j] + bv;
}

// M fixed = 64: A staged in LDS (async global->LDS), shared by all 8 waves.
// One wave per N-tile, B fragment reused across 4 M-tiles.
#define CHUNK_K 128
__global__ __launch_bounds__(256) void k_wmma_gemm_m64(
    const unsigned short* __restrict__ A_, const unsigned short* __restrict__ W_,
    const float* __restrict__ bias, float* __restrict__ C, int N, int K)
{
  const __bf16* W = (const __bf16*)W_;
  __shared__ __align__(16) unsigned short As[64*CHUNK_K];   // 16 KB
  int tid  = threadIdx.x;
  int wave = tid >> 5, lane = tid & 31;
  int tileN = blockIdx.x*8 + wave;
  bool active = tileN < (N >> 4);
  int l15   = lane & 15;
  int koffA = (lane >= 16) ? 16 : 0;
  int koffB = (lane >= 16) ? 16 : 0;
  const __bf16* wrow = active ? (W + (size_t)(tileN*16 + l15)*K) : W;
  v8f acc0={},acc1={},acc2={},acc3={};
  int sr  = tid >> 2;          // A row this thread stages (64 rows / 4 thr each)
  int seg = (tid & 3) * 32;    // 32 elems = 64 B segment within the chunk row

  for (int k0 = 0; k0 < K; k0 += CHUNK_K){
    __syncthreads();           // previous round's LDS reads done
    const unsigned short* gsrc = A_ + (size_t)sr*K + k0 + seg;
    unsigned short*       ldst = As + sr*CHUNK_K + seg;
#pragma unroll
    for (int q = 0; q < 4; q++)
      cp16_g2l(gsrc + q*8, ldst + q*8);          // 4 x b128 per thread
    async_wait0();
    __syncthreads();           // staged chunk visible to all waves
    if (active){
      const __bf16* Asb = (const __bf16*)As;
      __builtin_prefetch((const void*)(wrow + k0 + CHUNK_K), 0, 1);  // next W chunk
#pragma unroll
      for (int kk = 0; kk < CHUNK_K; kk += 32){
        v16bf b  = *(const v16bf*)(wrow + k0 + kk + koffB);
        v16bf a0 = load_a16x32(Asb, CHUNK_K,      l15, kk, koffA);
        v16bf a1 = load_a16x32(Asb, CHUNK_K, 16 + l15, kk, koffA);
        v16bf a2 = load_a16x32(Asb, CHUNK_K, 32 + l15, kk, koffA);
        v16bf a3 = load_a16x32(Asb, CHUNK_K, 48 + l15, kk, koffA);
        acc0 = __builtin_amdgcn_wmma_f32_16x16x32_bf16(false,a0,false,b,(short)0,acc0,false,false);
        acc1 = __builtin_amdgcn_wmma_f32_16x16x32_bf16(false,a1,false,b,(short)0,acc1,false,false);
        acc2 = __builtin_amdgcn_wmma_f32_16x16x32_bf16(false,a2,false,b,(short)0,acc2,false,false);
        acc3 = __builtin_amdgcn_wmma_f32_16x16x32_bf16(false,a3,false,b,(short)0,acc3,false,false);
      }
    }
  }
  if (active){
    int col = tileN*16 + l15;
    float bv = bias ? bias[col] : 0.f;
    int r0 = (lane>=16)?8:0;
#pragma unroll
    for (int j=0;j<8;j++){
      C[(size_t)(   r0+j)*N + col] = acc0[j] + bv;
      C[(size_t)(16+r0+j)*N + col] = acc1[j] + bv;
      C[(size_t)(32+r0+j)*N + col] = acc2[j] + bv;
      C[(size_t)(48+r0+j)*N + col] = acc3[j] + bv;
    }
  }
}

// --------------------------------------------------------------- GRU pieces
// r,z,n gates; h updated in place; bf16 copy for next GEMM; optional y write
__global__ void k_gru_gate(const float* __restrict__ gi, const float* __restrict__ gh,
                           float* __restrict__ h, unsigned short* __restrict__ hbf,
                           float* __restrict__ y, int ldy, int mode){
  int i = blockIdx.x*blockDim.x + threadIdx.x;
  if (i >= NB*NH) return;
  int b = i / NH, j = i % NH;
  const float* gib = gi + (size_t)b*G3;
  const float* ghb = gh + (size_t)b*G3;
  float r = sigmoidf_(gib[j]      + ghb[j]);
  float z = sigmoidf_(gib[NH+j]   + ghb[NH+j]);
  float n = tanhf    (gib[2*NH+j] + r*ghb[2*NH+j]);
  float h2 = (1.f - z)*n + z*h[i];
  h[i]   = h2;
  hbf[i] = f32_to_bf16(h2);
  if (mode == 0)      y[(size_t)b*ldy + j]  = h2;
  else if (mode == 1) y[(size_t)b*ldy + j] += h2;
}

__global__ void k_add_h(const float* __restrict__ a, const float* __restrict__ b_,
                        float* __restrict__ h, unsigned short* __restrict__ hbf){
  int i = blockIdx.x*blockDim.x + threadIdx.x;
  if (i >= NB*NH) return;
  float v = a[i] + b_[i];
  h[i] = v;
  hbf[i] = f32_to_bf16(v);
}

// dot-product attention over Te=40; writes cat = [h2 | ctx] in bf16
__global__ __launch_bounds__(256) void k_attn(const float* __restrict__ y_enc,
                                              const float* __restrict__ h2,
                                              unsigned short* __restrict__ cat){
  int b = blockIdx.x;
  int lane = threadIdx.x & 31, w = threadIdx.x >> 5;
  __shared__ float sc[TE];
  __shared__ float sinv;
  const float* hb = h2 + (size_t)b*NH;
  for (int t = w; t < TE; t += 8){
    const float* yr = y_enc + ((size_t)b*TE + t)*NH;
    float p = 0.f;
    for (int j = lane; j < NH; j += 32) p += yr[j]*hb[j];
    for (int o=16;o;o>>=1) p += __shfl_down(p,o);
    if (lane==0) sc[t] = p;
  }
  __syncthreads();
  if (threadIdx.x == 0){
    float m = -1e30f;
    for (int t=0;t<TE;t++) m = fmaxf(m, sc[t]);
    float s = 0.f;
    for (int t=0;t<TE;t++){ float e = __expf(sc[t]-m); sc[t]=e; s+=e; }
    sinv = 1.f/s;
  }
  __syncthreads();
  float inv = sinv;
  for (int j = threadIdx.x; j < NH; j += 256){
    float acc = 0.f;
    for (int t=0;t<TE;t++) acc += sc[t]*y_enc[((size_t)b*TE+t)*NH + j];
    cat[(size_t)b*HH2 + NH + j] = f32_to_bf16(acc*inv);
    cat[(size_t)b*HH2 + j]      = f32_to_bf16(hb[j]);
  }
}

// softmax over V=30000, writes out[b, t, :]
__global__ __launch_bounds__(512) void k_softmax_out(const float* __restrict__ logits,
                                                     float* __restrict__ out, int t){
  int b = blockIdx.x;
  const float* row = logits + (size_t)b*NV;
  __shared__ float sred[16];
  int lane = threadIdx.x & 31, w = threadIdx.x >> 5;
  float m = -1e30f;
  for (int v = threadIdx.x; v < NV; v += 512) m = fmaxf(m, row[v]);
  for (int o=16;o;o>>=1) m = fmaxf(m, __shfl_down(m,o));
  if (lane==0) sred[w] = m;
  __syncthreads();
  if (w==0){
    float v2 = (lane<16) ? sred[lane] : -1e30f;
    for (int o=8;o;o>>=1) v2 = fmaxf(v2, __shfl_down(v2,o));
    if (lane==0) sred[0] = v2;
  }
  __syncthreads();
  m = sred[0];
  __syncthreads();
  float s = 0.f;
  for (int v = threadIdx.x; v < NV; v += 512) s += __expf(row[v]-m);
  for (int o=16;o;o>>=1) s += __shfl_down(s,o);
  if (lane==0) sred[w] = s;
  __syncthreads();
  if (w==0){
    float v2 = (lane<16) ? sred[lane] : 0.f;
    for (int o=8;o;o>>=1) v2 += __shfl_down(v2,o);
    if (lane==0) sred[0] = v2;
  }
  __syncthreads();
  float inv = 1.f/sred[0];
  float* o = out + ((size_t)b*TDD + t)*NV;
  for (int v = threadIdx.x; v < NV; v += 512) o[v] = __expf(row[v]-m)*inv;
}

// ------------------------------------------------------------------- driver
extern "C" void kernel_launch(void* const* d_in, const int* in_sizes, int n_in,
                              void* d_out, int out_size, void* d_ws, size_t ws_size,
                              hipStream_t stream) {
  (void)in_sizes; (void)n_in; (void)out_size; (void)ws_size;
  const int*   x_enc   = (const int*)d_in[0];
  const int*   x_dec   = (const int*)d_in[1];
  const float* emb_enc = (const float*)d_in[2];
  const float* Wih_f   = (const float*)d_in[3];
  const float* Whh_f   = (const float*)d_in[4];
  const float* bih_f   = (const float*)d_in[5];
  const float* bhh_f   = (const float*)d_in[6];
  const float* Wih_b   = (const float*)d_in[7];
  const float* Whh_b   = (const float*)d_in[8];
  const float* bih_b   = (const float*)d_in[9];
  const float* bhh_b   = (const float*)d_in[10];
  const float* emb_dec = (const float*)d_in[11];
  const float* Wih_d   = (const float*)d_in[12];
  const float* Whh_d   = (const float*)d_in[13];
  const float* bih_d   = (const float*)d_in[14];
  const float* bhh_d   = (const float*)d_in[15];
  const float* fc_W    = (const float*)d_in[16];
  const float* fc_b    = (const float*)d_in[17];
  float* out = (float*)d_out;

  char* ws = (char*)d_ws;
  size_t off = 0;
  auto wsalloc = [&](size_t bytes)->void*{
    void* p = ws + off;
    off = (off + bytes + 255) & ~(size_t)255;
    return p;
  };
  unsigned short* wihf  = (unsigned short*)wsalloc((size_t)G3*NEP*2);
  unsigned short* wihb  = (unsigned short*)wsalloc((size_t)G3*NEP*2);
  unsigned short* wihd  = (unsigned short*)wsalloc((size_t)G3*NEP*2);
  unsigned short* whhf  = (unsigned short*)wsalloc((size_t)G3*NH*2);
  unsigned short* whhb  = (unsigned short*)wsalloc((size_t)G3*NH*2);
  unsigned short* whhd  = (unsigned short*)wsalloc((size_t)G3*NH*2);
  unsigned short* fcw   = (unsigned short*)wsalloc((size_t)NV*HH2*2);
  unsigned short* xse   = (unsigned short*)wsalloc((size_t)TE*NB*NEP*2);
  unsigned short* xsd   = (unsigned short*)wsalloc((size_t)TDD*NB*NEP*2);
  float* gi_f  = (float*)wsalloc((size_t)TE*NB*G3*4);
  float* gi_b  = (float*)wsalloc((size_t)TE*NB*G3*4);
  float* gi_d  = (float*)wsalloc((size_t)TDD*NB*G3*4);
  float* gh    = (float*)wsalloc((size_t)NB*G3*4);
  float* y_enc = (float*)wsalloc((size_t)NB*TE*NH*4);
  float* h_f   = (float*)wsalloc((size_t)NB*NH*4);
  float* h_b   = (float*)wsalloc((size_t)NB*NH*4);
  float* h_d   = (float*)wsalloc((size_t)NB*NH*4);
  unsigned short* hf_bf = (unsigned short*)wsalloc((size_t)NB*NH*2);
  unsigned short* hb_bf = (unsigned short*)wsalloc((size_t)NB*NH*2);
  unsigned short* hd_bf = (unsigned short*)wsalloc((size_t)NB*NH*2);
  unsigned short* cat   = (unsigned short*)wsalloc((size_t)NB*HH2*2);
  float* logits = (float*)wsalloc((size_t)NB*NV*4);

  auto cdiv = [](long long a, long long b){ return (int)((a + b - 1)/b); };

  // --- weight / input conversion to bf16 (fc_W: 107.5MB bf16 -> L2 resident)
  k_conv_bf16<<<cdiv((long long)G3*NEP,256),256,0,stream>>>(Wih_f, wihf, G3, NE, NEP);
  k_conv_bf16<<<cdiv((long long)G3*NEP,256),256,0,stream>>>(Wih_b, wihb, G3, NE, NEP);
  k_conv_bf16<<<cdiv((long long)G3*NEP,256),256,0,stream>>>(Wih_d, wihd, G3, NE, NEP);
  k_conv_bf16<<<cdiv((long long)G3*NH,256),256,0,stream>>>(Whh_f, whhf, G3, NH, NH);
  k_conv_bf16<<<cdiv((long long)G3*NH,256),256,0,stream>>>(Whh_b, whhb, G3, NH, NH);
  k_conv_bf16<<<cdiv((long long)G3*NH,256),256,0,stream>>>(Whh_d, whhd, G3, NH, NH);
  k_conv_bf16<<<cdiv((long long)NV*HH2,256),256,0,stream>>>(fc_W, fcw, NV, HH2, HH2);
  k_gather_emb<<<cdiv((long long)TE*NB*NEP,256),256,0,stream>>>(emb_enc, x_enc, xse, TE);
  k_gather_emb<<<cdiv((long long)TDD*NB*NEP,256),256,0,stream>>>(emb_dec, x_dec, xsd, TDD);

  // --- hoisted input-side gate GEMMs: gi = xs @ Wih^T + bih  (M = T*B = 2560)
  int giBlocks = cdiv((long long)(TE*NB/16)*(G3/16), 8);
  k_wmma_gemm_tile<<<giBlocks,256,0,stream>>>(xse, wihf, bih_f, gi_f, TE*NB, G3, NEP);
  k_wmma_gemm_tile<<<giBlocks,256,0,stream>>>(xse, wihb, bih_b, gi_b, TE*NB, G3, NEP);
  k_wmma_gemm_tile<<<giBlocks,256,0,stream>>>(xsd, wihd, bih_d, gi_d, TDD*NB, G3, NEP);

  // --- init hidden states
  k_zero_f32<<<cdiv(NB*NH,256),256,0,stream>>>(h_f, NB*NH);
  k_zero_f32<<<cdiv(NB*NH,256),256,0,stream>>>(h_b, NB*NH);
  k_zero_f32<<<cdiv(NB*NH/2,256),256,0,stream>>>((float*)hf_bf, NB*NH/2);
  k_zero_f32<<<cdiv(NB*NH/2,256),256,0,stream>>>((float*)hb_bf, NB*NH/2);

  int ghBlocks   = cdiv((long long)(NB/16)*(G3/16), 8);   // 84
  int gateBlocks = cdiv(NB*NH,256);                       // 224

  // --- encoder forward
  for (int t = 0; t < TE; t++){
    k_wmma_gemm_tile<<<ghBlocks,256,0,stream>>>(hf_bf, whhf, bhh_f, gh, NB, G3, NH);
    k_gru_gate<<<gateBlocks,256,0,stream>>>(gi_f + (size_t)t*NB*G3, gh, h_f, hf_bf,
                                            y_enc + (size_t)t*NH, TE*NH, /*store*/0);
  }
  // --- encoder backward
  for (int t = TE-1; t >= 0; t--){
    k_wmma_gemm_tile<<<ghBlocks,256,0,stream>>>(hb_bf, whhb, bhh_b, gh, NB, G3, NH);
    k_gru_gate<<<gateBlocks,256,0,stream>>>(gi_b + (size_t)t*NB*G3, gh, h_b, hb_bf,
                                            y_enc + (size_t)t*NH, TE*NH, /*add*/1);
  }
  // --- decoder initial hidden = h_f + h_b
  k_add_h<<<gateBlocks,256,0,stream>>>(h_f, h_b, h_d, hd_bf);

  // --- decoder steps
  int fcBlocks = cdiv(NV/16, 8);   // 235 blocks * 8 waves >= 1875 tiles
  for (int t = 0; t < TDD; t++){
    k_wmma_gemm_tile<<<ghBlocks,256,0,stream>>>(hd_bf, whhd, bhh_d, gh, NB, G3, NH);
    k_gru_gate<<<gateBlocks,256,0,stream>>>(gi_d + (size_t)t*NB*G3, gh, h_d, hd_bf,
                                            nullptr, 0, /*none*/2);
    k_attn<<<NB,256,0,stream>>>(y_enc, h_d, cat);
    k_wmma_gemm_m64<<<fcBlocks,256,0,stream>>>(cat, fcw, fc_b, logits, NV, HH2);
    k_softmax_out<<<NB,512,0,stream>>>(logits, out, t);
  }
}
